// EquivariantAttention_16423954939932
// MI455X (gfx1250) — matrix operand order repaired and checked by reading, not compiled, not run on hardware
//
#include <hip/hip_runtime.h>
#include <hip/hip_bf16.h>

// ---------------------------------------------------------------------------
// EquivariantAttention on gfx1250 (MI455X): bf16 WMMA flash-attention pipeline
//   K1: qkv = x @ qkv_w^T + b  -> Qbf, Kbf (row major), VTbf (transposed)
//   K2: flash attention with distance-modulated scores -> AObf
//   K3: out = AO @ out_w^T + b, masked -> fp32 d_out
// Workspace: 4 x 16MB bf16 planes = 64MB.
// ---------------------------------------------------------------------------

#define B_ 8
#define N_ 2048
#define H_ 512
#define BC 64
#define SCALE 0.04419417382415922f   // 1/sqrt(512)

typedef __attribute__((ext_vector_type(16))) __bf16 v16bf;
typedef __attribute__((ext_vector_type(8)))  __bf16 v8bf;
typedef __attribute__((ext_vector_type(8)))  float  v8f;

#define WMMA_BF16(A, B, C) \
  __builtin_amdgcn_wmma_f32_16x16x32_bf16(false, (A), false, (B), (short)0, (C), false, false)

// --- CDNA5 async global->LDS DMA path (ASYNCcnt-tracked) -------------------
#if defined(__has_builtin)
#if __has_builtin(__builtin_amdgcn_global_load_async_to_lds_b128) && \
    __has_builtin(__builtin_amdgcn_s_wait_asynccnt)
#define USE_ASYNC_LDS 1
#endif
#endif
#ifndef USE_ASYNC_LDS
#define USE_ASYNC_LDS 0
#endif

#if USE_ASYNC_LDS
typedef int v4i_t __attribute__((__vector_size__(4 * sizeof(int))));
typedef __attribute__((address_space(1))) v4i_t as1_v4i;
typedef __attribute__((address_space(3))) v4i_t as3_v4i;
#define ASYNC_CP16(gp, lp) \
  __builtin_amdgcn_global_load_async_to_lds_b128((as1_v4i*)(gp), (as3_v4i*)(lp), 0, 0)
#define ASYNC_WAIT() __builtin_amdgcn_s_wait_asynccnt(0)
#endif

static __device__ __forceinline__ v8f vzero8() {
  v8f z;
#pragma unroll
  for (int i = 0; i < 8; ++i) z[i] = 0.0f;
  return z;
}

static __device__ __forceinline__ v16bf cat16(v8bf lo, v8bf hi) {
  return __builtin_shufflevector(lo, hi, 0, 1, 2, 3, 4, 5, 6, 7, 8, 9, 10, 11, 12, 13, 14, 15);
}

// A fragment (16x32 bf16) from row-major fp32: 8 elems at p, 8 at p+16
static __device__ __forceinline__ v16bf load_a_f32(const float* p) {
  v16bf r;
#pragma unroll
  for (int i = 0; i < 8; ++i) r[i] = (__bf16)p[i];
#pragma unroll
  for (int i = 0; i < 8; ++i) r[i + 8] = (__bf16)p[16 + i];
  return r;
}
// B fragment (32x16 bf16) from fp32: 16 contiguous K values of one column-row
static __device__ __forceinline__ v16bf load_b_f32(const float* p) {
  v16bf r;
#pragma unroll
  for (int i = 0; i < 16; ++i) r[i] = (__bf16)p[i];
  return r;
}
// A fragment from bf16 memory (global or LDS)
static __device__ __forceinline__ v16bf load_a_bf(const __bf16* p) {
  v8bf lo = *(const v8bf*)p;
  v8bf hi = *(const v8bf*)(p + 16);
  return cat16(lo, hi);
}

// ---------------------------------------------------------------------------
// Kernel 1: QKV projection. 2x2 tiles (32x32 output) per wave so each A/B
// fragment feeds two WMMAs. K=512 -> 16 k-steps, 64 WMMAs per wave.
// cols e: [0,512)->Q, [512,1024)->K, [1024,1536)->V stored transposed (B,H,N).
// ---------------------------------------------------------------------------
__global__ void __launch_bounds__(256)
qkv_proj_kernel(const float* __restrict__ x, const float* __restrict__ w,
                const float* __restrict__ bias,
                __bf16* __restrict__ Qbf, __bf16* __restrict__ Kbf,
                __bf16* __restrict__ VTbf) {
  const int wave = threadIdx.x >> 5;
  const int lane = threadIdx.x & 31;
  const int lrow = lane & 15;
  const int lhi  = lane >> 4;
  const int E_T2 = (3 * H_) / 32;           // 48 column super-tiles
  const int tile = blockIdx.x * 8 + wave;
  const int tm = tile / E_T2, te = tile % E_T2;
  const int m0 = tm * 32, e0 = te * 32;

  const float* arow0 = x + (size_t)(m0 + lrow) * H_;
  const float* arow1 = arow0 + (size_t)16 * H_;
  const float* brow0 = w + (size_t)(e0 + lrow) * H_;
  const float* brow1 = brow0 + (size_t)16 * H_;

  v8f acc00 = vzero8(), acc01 = vzero8(), acc10 = vzero8(), acc11 = vzero8();
#pragma unroll
  for (int k0 = 0; k0 < H_; k0 += 32) {
    const int ao = k0 + (lhi << 3), bo = k0 + (lhi << 3);
    v16bf a0 = load_a_f32(arow0 + ao);
    v16bf a1 = load_a_f32(arow1 + ao);
    v16bf b0 = load_a_f32(brow0 + bo);
    v16bf b1 = load_a_f32(brow1 + bo);
    acc00 = WMMA_BF16(a0, b0, acc00);
    acc01 = WMMA_BF16(a0, b1, acc01);
    acc10 = WMMA_BF16(a1, b0, acc10);
    acc11 = WMMA_BF16(a1, b1, acc11);
  }

  const int rbase = lhi << 3;
#pragma unroll
  for (int c = 0; c < 2; ++c) {
    const int e = e0 + c * 16 + lrow;       // column, fixed per lane
    const float bv = bias[e];
#pragma unroll
    for (int rt = 0; rt < 2; ++rt) {
      const v8f acc = (c == 0) ? (rt == 0 ? acc00 : acc10)
                               : (rt == 0 ? acc01 : acc11);
#pragma unroll
      for (int r = 0; r < 8; ++r) {
        const int m = m0 + rt * 16 + rbase + r;
        const __bf16 h = (__bf16)(acc[r] + bv);
        if (e < 512) {
          Qbf[(size_t)m * H_ + e] = h;
        } else if (e < 1024) {
          Kbf[(size_t)m * H_ + (e - 512)] = h;
        } else {
          const int bb = m >> 11, n = m & (N_ - 1);
          VTbf[((size_t)bb * H_ + (e - 1024)) * N_ + n] = h;
        }
      }
    }
  }
}

// ---------------------------------------------------------------------------
// Kernel 2: flash attention with distance-modulated scores.
// 128 threads (4 waves), 16 queries/block, 64-key chunks, online softmax.
// Per wave: 16x16 S tile (its 16 keys), 16x128 fp32 O accumulator (h slice).
// K/V^T chunks staged via async global->LDS DMA when available.
// ---------------------------------------------------------------------------
constexpr int SMEM_BF16_ELEMS = 16 * H_ + BC * H_ + H_ * BC + 16 * BC; // Qs,Ks,VTs,Ps
constexpr int SMEM_FLOATS = 16 * 4 /*m,l,mnew,f*/ + 64 * 2 /*wmax,wsum*/ +
                            BC * 4 /*gk xyz,sq*/ + 16 * 4 /*gq xyz,sq*/;
constexpr int SMEM_BYTES = SMEM_BF16_ELEMS * 2 + SMEM_FLOATS * 4 + BC /*mask*/;

__global__ void __launch_bounds__(128)
attn_flash_kernel(const float* __restrict__ g, const unsigned char* __restrict__ mask,
                  const __bf16* __restrict__ Qbf, const __bf16* __restrict__ Kbf,
                  const __bf16* __restrict__ VTbf, __bf16* __restrict__ AObf) {
  extern __shared__ char smem[];
  __bf16* Qs  = (__bf16*)smem;              // [16][512]
  __bf16* Ks  = Qs + 16 * H_;               // [64][512]
  __bf16* VTs = Ks + BC * H_;               // [512][64]
  __bf16* Ps  = VTs + H_ * BC;              // [16][64]
  float* m_run = (float*)(Ps + 16 * BC);    // [16]
  float* l_run = m_run + 16;                // [16]
  float* mnew  = l_run + 16;                // [16]
  float* fbuf  = mnew + 16;                 // [16]
  float* wmax  = fbuf + 16;                 // [4][16]
  float* wsum  = wmax + 64;                 // [4][16]
  float* gkx   = wsum + 64;                 // [64] x4
  float* gky   = gkx + BC;
  float* gkz   = gky + BC;
  float* sqk   = gkz + BC;
  float* gqx   = sqk + BC;                  // [16] x4
  float* gqy   = gqx + 16;
  float* gqz   = gqy + 16;
  float* sqq   = gqz + 16;
  unsigned char* mk = (unsigned char*)(sqq + 16); // [64]

  const int tid  = threadIdx.x;
  const int wave = tid >> 5;
  const int lane = tid & 31;
  const int lrow = lane & 15;
  const int lhi  = lane >> 4;
  const int rbase = lhi << 3;

  const int b  = blockIdx.x >> 7;           // / 128 query tiles per batch
  const int i0 = (blockIdx.x & 127) << 4;

  { // stage Q tile: 16KB contiguous
    const __bf16* src = Qbf + (size_t)(b * N_ + i0) * H_;
#if USE_ASYNC_LDS
    for (int idx = tid; idx < 16 * H_ / 8; idx += 128)
      ASYNC_CP16(src + idx * 8, Qs + idx * 8);
#else
    for (int idx = tid; idx < 16 * H_ / 8; idx += 128)
      ((uint4*)Qs)[idx] = ((const uint4*)src)[idx];
#endif
  }
  if (tid < 16) {
    const float* gp = g + (size_t)(b * N_ + i0 + tid) * 3;
    const float X = gp[0], Y = gp[1], Z = gp[2];
    gqx[tid] = X; gqy[tid] = Y; gqz[tid] = Z;
    sqq[tid] = X * X + Y * Y + Z * Z;
    m_run[tid] = -INFINITY;
    l_run[tid] = 0.0f;
  }
#if USE_ASYNC_LDS
  ASYNC_WAIT();
#endif
  __syncthreads();

  // hoist this lane's 8 query-row geometry values into registers
  float gqx_r[8], gqy_r[8], gqz_r[8], sqq_r[8];
#pragma unroll
  for (int r = 0; r < 8; ++r) {
    const int row = rbase + r;
    gqx_r[r] = gqx[row]; gqy_r[r] = gqy[row]; gqz_r[r] = gqz[row]; sqq_r[r] = sqq[row];
  }

  v8f oacc[8];
#pragma unroll
  for (int t = 0; t < 8; ++t) oacc[t] = vzero8();

  const int jl = (wave << 4) + lrow;        // this lane's key column 0..63

  for (int it = 0; it < N_ / BC; ++it) {
    const int j0 = it * BC;
    __syncthreads();                        // prior-iter LDS reads done
    { // stage K chunk (64KB contiguous) and V^T chunk (512 rows x 64 cols)
      const __bf16* ksrc  = Kbf + (size_t)(b * N_ + j0) * H_;
      const __bf16* vbase = VTbf + (size_t)b * H_ * N_ + j0;
#if USE_ASYNC_LDS
      for (int idx = tid; idx < BC * H_ / 8; idx += 128)
        ASYNC_CP16(ksrc + idx * 8, Ks + idx * 8);
      for (int idx = tid; idx < H_ * BC / 8; idx += 128) {
        const int h = idx >> 3, p = idx & 7;
        ASYNC_CP16(vbase + (size_t)h * N_ + p * 8, VTs + idx * 8);
      }
#else
      for (int idx = tid; idx < BC * H_ / 8; idx += 128)
        ((uint4*)Ks)[idx] = ((const uint4*)ksrc)[idx];
      for (int idx = tid; idx < H_ * BC / 8; idx += 128) {
        const int h = idx >> 3, p = idx & 7;
        ((uint4*)VTs)[idx] = *(const uint4*)(vbase + (size_t)h * N_ + p * 8);
      }
#endif
    }
    if (tid < BC) {
      const float* gp = g + (size_t)(b * N_ + j0 + tid) * 3;
      const float X = gp[0], Y = gp[1], Z = gp[2];
      gkx[tid] = X; gky[tid] = Y; gkz[tid] = Z;
      sqk[tid] = X * X + Y * Y + Z * Z;
      mk[tid] = mask[(size_t)b * N_ + j0 + tid];
    }
#if USE_ASYNC_LDS
    ASYNC_WAIT();
#endif
    __syncthreads();

    // ---- S = (Q K^T) * scale, 16x16 tile per wave ----
    v8f sacc = vzero8();
#pragma unroll
    for (int kk = 0; kk < H_ / 32; ++kk) {
      const int k0 = kk * 32;
      const __bf16* ap = Qs + lrow * H_ + k0 + (lhi << 3);
      v16bf afrag = cat16(*(const v8bf*)ap, *(const v8bf*)(ap + 16));
      v16bf bfrag = load_a_bf(Ks + jl * H_ + k0 + (lhi << 3));
      sacc = WMMA_BF16(afrag, bfrag, sacc);
    }

    // distance modulation + key mask on C fragment
    const float gjx = gkx[jl], gjy = gky[jl], gjz = gkz[jl], sj = sqk[jl];
    const bool mok = (mk[jl] != 0);
    float svals[8];
#pragma unroll
    for (int r = 0; r < 8; ++r) {
      float s = sacc[r] * SCALE;
      const float d2 = sqq_r[r] + sj -
                       2.0f * (gqx_r[r] * gjx + gqy_r[r] * gjy + gqz_r[r] * gjz);
      s *= __expf(-sqrtf(fmaxf(d2, 0.0f)));
      svals[r] = mok ? s : -INFINITY;
    }

    // per-tile row max: reduce across the 16 lanes of each half-wave
    float red[8];
#pragma unroll
    for (int r = 0; r < 8; ++r) red[r] = svals[r];
#pragma unroll
    for (int mm = 8; mm >= 1; mm >>= 1)
#pragma unroll
      for (int r = 0; r < 8; ++r)
        red[r] = fmaxf(red[r], __shfl_xor(red[r], mm, 32));
    if (lrow == 0) {
#pragma unroll
      for (int r = 0; r < 8; ++r) wmax[(wave << 4) + rbase + r] = red[r];
    }
    __syncthreads();
    if (tid < 16) {  // online-softmax stats for query row `tid`
      const float mo = m_run[tid];
      const float mc = fmaxf(fmaxf(wmax[tid], wmax[16 + tid]),
                             fmaxf(wmax[32 + tid], wmax[48 + tid]));
      const float mn = fmaxf(mo, mc);
      mnew[tid] = mn;
      fbuf[tid] = (mo == -INFINITY) ? 0.0f : __expf(mo - mn);
      m_run[tid] = mn;
    }
    __syncthreads();

    // P = exp(S - m_new) -> bf16 LDS tile + row sums
#pragma unroll
    for (int r = 0; r < 8; ++r) {
      const int row = rbase + r;
      const float p = __expf(svals[r] - mnew[row]);
      Ps[row * BC + jl] = (__bf16)p;
      red[r] = p;
    }
#pragma unroll
    for (int mm = 8; mm >= 1; mm >>= 1)
#pragma unroll
      for (int r = 0; r < 8; ++r)
        red[r] += __shfl_xor(red[r], mm, 32);
    if (lrow == 0) {
#pragma unroll
      for (int r = 0; r < 8; ++r) wsum[(wave << 4) + rbase + r] = red[r];
    }

    // rescale running O by exp(m_old - m_new)
    float fr[8];
#pragma unroll
    for (int r = 0; r < 8; ++r) fr[r] = fbuf[rbase + r];
#pragma unroll
    for (int t = 0; t < 8; ++t)
#pragma unroll
      for (int r = 0; r < 8; ++r) oacc[t][r] *= fr[r];
    __syncthreads();                        // Ps & wsum complete
    if (tid < 16)
      l_run[tid] = l_run[tid] * fbuf[tid] +
                   wsum[tid] + wsum[16 + tid] + wsum[32 + tid] + wsum[48 + tid];

    // ---- O += P @ V over this wave's 128-wide h slice ----
#pragma unroll
    for (int t = 0; t < 8; ++t) {
      const int h = (wave << 7) + (t << 4) + lrow;
#pragma unroll
      for (int kk = 0; kk < 2; ++kk) {
        const __bf16* ap = Ps + lrow * BC + kk * 32 + (lhi << 3);
        v16bf afrag = cat16(*(const v8bf*)ap, *(const v8bf*)(ap + 16));
        v16bf bfrag = load_a_bf(VTs + h * BC + kk * 32 + (lhi << 3));
        oacc[t] = WMMA_BF16(afrag, bfrag, oacc[t]);
      }
    }
  }

  __syncthreads();
  float linv[8];
#pragma unroll
  for (int r = 0; r < 8; ++r) linv[r] = 1.0f / fmaxf(l_run[rbase + r], 1e-30f);
#pragma unroll
  for (int t = 0; t < 8; ++t) {
    const int h = (wave << 7) + (t << 4) + lrow;
#pragma unroll
    for (int r = 0; r < 8; ++r)
      AObf[(size_t)(b * N_ + i0 + rbase + r) * H_ + h] = (__bf16)(oacc[t][r] * linv[r]);
  }
}

// ---------------------------------------------------------------------------
// Kernel 3: output projection + bias + query mask, fp32 out. 2x2 wave tiles.
// ---------------------------------------------------------------------------
__global__ void __launch_bounds__(256)
out_proj_kernel(const __bf16* __restrict__ AObf, const float* __restrict__ w,
                const float* __restrict__ bias, const unsigned char* __restrict__ mask,
                float* __restrict__ out) {
  const int wave = threadIdx.x >> 5;
  const int lane = threadIdx.x & 31;
  const int lrow = lane & 15;
  const int lhi  = lane >> 4;
  const int O_T2 = H_ / 32;                 // 16 column super-tiles
  const int tile = blockIdx.x * 8 + wave;
  const int tm = tile / O_T2, to = tile % O_T2;
  const int m0 = tm * 32, o0 = to * 32;

  const __bf16* arow0 = AObf + (size_t)(m0 + lrow) * H_;
  const __bf16* arow1 = arow0 + (size_t)16 * H_;
  const float*  brow0 = w + (size_t)(o0 + lrow) * H_;
  const float*  brow1 = brow0 + (size_t)16 * H_;

  v8f acc00 = vzero8(), acc01 = vzero8(), acc10 = vzero8(), acc11 = vzero8();
#pragma unroll
  for (int k0 = 0; k0 < H_; k0 += 32) {
    const int ao = k0 + (lhi << 3), bo = k0 + (lhi << 3);
    v16bf a0 = load_a_bf(arow0 + ao);
    v16bf a1 = load_a_bf(arow1 + ao);
    v16bf b0 = load_a_f32(brow0 + bo);
    v16bf b1 = load_a_f32(brow1 + bo);
    acc00 = WMMA_BF16(a0, b0, acc00);
    acc01 = WMMA_BF16(a0, b1, acc01);
    acc10 = WMMA_BF16(a1, b0, acc10);
    acc11 = WMMA_BF16(a1, b1, acc11);
  }

  const int rbase = lhi << 3;
#pragma unroll
  for (int c = 0; c < 2; ++c) {
    const int o = o0 + c * 16 + lrow;
    const float bv = bias[o];
#pragma unroll
    for (int rt = 0; rt < 2; ++rt) {
      const v8f acc = (c == 0) ? (rt == 0 ? acc00 : acc10)
                               : (rt == 0 ? acc01 : acc11);
#pragma unroll
      for (int r = 0; r < 8; ++r) {
        const int m = m0 + rt * 16 + rbase + r;
        const float mv = mask[m] ? 1.0f : 0.0f;
        out[(size_t)m * H_ + o] = (acc[r] + bv) * mv;
      }
    }
  }
}

// ---------------------------------------------------------------------------
extern "C" void kernel_launch(void* const* d_in, const int* in_sizes, int n_in,
                              void* d_out, int out_size, void* d_ws, size_t ws_size,
                              hipStream_t stream) {
  const float* x          = (const float*)d_in[0];
  const float* g          = (const float*)d_in[1];
  const unsigned char* mk = (const unsigned char*)d_in[2];
  const float* qkv_w      = (const float*)d_in[3];
  const float* qkv_b      = (const float*)d_in[4];
  const float* out_w      = (const float*)d_in[5];
  const float* out_b      = (const float*)d_in[6];
  float* out = (float*)d_out;
  (void)in_sizes; (void)n_in; (void)out_size; (void)ws_size;

  char* ws = (char*)d_ws;
  const size_t PLANE = (size_t)B_ * N_ * H_ * 2; // 16 MB per bf16 plane
  __bf16* Qbf  = (__bf16*)(ws);
  __bf16* Kbf  = (__bf16*)(ws + PLANE);
  __bf16* VTbf = (__bf16*)(ws + 2 * PLANE);
  __bf16* AObf = (__bf16*)(ws + 3 * PLANE);

  (void)hipFuncSetAttribute((const void*)attn_flash_kernel,
                            hipFuncAttributeMaxDynamicSharedMemorySize, SMEM_BYTES);

  {
    const int tiles = (B_ * N_ / 32) * (3 * H_ / 32);    // 512*48 = 24576
    qkv_proj_kernel<<<dim3(tiles / 8), dim3(256), 0, stream>>>(x, qkv_w, qkv_b,
                                                               Qbf, Kbf, VTbf);
  }
  attn_flash_kernel<<<dim3(B_ * N_ / 16), dim3(128), SMEM_BYTES, stream>>>(
      g, mk, Qbf, Kbf, VTbf, AObf);
  {
    const int tiles = (B_ * N_ / 32) * (H_ / 32);        // 512*16 = 8192
    out_proj_kernel<<<dim3(tiles / 8), dim3(256), 0, stream>>>(AObf, out_w, out_b,
                                                               mk, out);
  }
}
